// TPUOptimizedAttention_57690000720292
// MI455X (gfx1250) — matrix operand order repaired and checked by reading, not compiled, not run on hardware
//
#include <hip/hip_runtime.h>
#include <hip/hip_bf16.h>

// ---------------------------------------------------------------------------
// Fused multi-head attention for MI455X (gfx1250), bf16 WMMA + TDM pipeline.
//   B=4, S=2048, D=2048, H=16, hd=128
// Pipeline:
//   1) f32 -> bf16 convert (x, w_qkv, w_o)
//   2) GEMM  qkv = x @ w_qkv^T + b_qkv   -> scatter bf16 Q,K,V [B,H,S,hd]
//      (LDS tiles staged by Tensor Data Mover, double-buffered)
//   3) flash-attention per (b,h,qtile)   -> bf16 attn_out [B,S,D]
//      (K tile staged by TDM overlapping manual V-transpose staging)
//   4) GEMM  out = attn_out @ w_o^T + b_o -> f32 d_out [B,S,D]
// ---------------------------------------------------------------------------

typedef __attribute__((ext_vector_type(16))) __bf16 v16bf;
typedef __attribute__((ext_vector_type(8)))  float  v8f;

union FragBF {            // one WMMA bf16 operand = 16 bf16 = 2 x b128
  v16bf v;
  uint4 q[2];
};

__device__ __forceinline__ unsigned short f2bf(float f) {
  union { float f; unsigned u; } x; x.f = f;
  unsigned r = x.u + 0x7FFFu + ((x.u >> 16) & 1u);   // round-to-nearest-even
  return (unsigned short)(r >> 16);
}

// ---------------------------------------------------------------------------
// Tensor Data Mover (gfx1250): 2D bf16 tile  global -> LDS.
// D# layout per CDNA5 ISA (group0 + group1, 2D tensor,
// groups 2/3 zero).  6-arg builtin form (clang-23 / therock-10.0 headers):
//   (uint32x4 g0, int32x8 g1, int32x4 g2, int32x4 g3, int32x8 pad, i32 cpol)
// ---------------------------------------------------------------------------
#if defined(__HIP_DEVICE_COMPILE__) && \
    __has_builtin(__builtin_amdgcn_tensor_load_to_lds) && \
    __has_builtin(__builtin_amdgcn_s_wait_tensorcnt)
#define HAVE_TDM 1
#else
#define HAVE_TDM 0
#endif

#if HAVE_TDM
typedef __attribute__((ext_vector_type(4))) unsigned int tdm_u32x4;
typedef __attribute__((ext_vector_type(8))) int          tdm_i32x8;
typedef __attribute__((ext_vector_type(4))) int          tdm_i32x4;

// tile_w elements per row, tile_h rows; row pitch = stride elements (bf16).
__device__ __forceinline__ void tdm_load_2d_bf16(unsigned lds_off,
                                                 const unsigned short* gptr,
                                                 unsigned tile_w,
                                                 unsigned tile_h,
                                                 unsigned stride) {
  const unsigned long long ga = (unsigned long long)(size_t)gptr;
  tdm_u32x4 g0;
  g0[0] = 1u;                                    // count=1, user descriptor
  g0[1] = lds_off;                               // lds_addr [63:32]
  g0[2] = (unsigned)(ga & 0xFFFFFFFFu);          // global_addr [95:64]
  g0[3] = (unsigned)((ga >> 32) & 0x01FFFFFFu)   // global_addr [120:96]
          | 0x80000000u;                         // type=2 ("image") [127:126]
  tdm_i32x8 g1;
  g1[0] = (int)(1u << 16);                       // data_size=1 (2 bytes)
  g1[1] = (int)((tile_w & 0xFFFFu) << 16);       // tensor_dim0 lo16
  g1[2] = (int)((tile_w >> 16) | ((tile_h & 0xFFFFu) << 16)); // d0 hi | d1 lo
  g1[3] = (int)((tile_h >> 16) | (tile_w << 16));             // d1 hi | tile_dim0
  g1[4] = (int)(tile_h & 0xFFFFu);               // tile_dim1 (tile_dim2=0)
  g1[5] = (int)stride;                           // tensor_dim0_stride lo32
  g1[6] = 0;                                     // stride hi16 | dim1_stride lo
  g1[7] = 0;
  const tdm_i32x4 z4 = {0, 0, 0, 0};
  const tdm_i32x8 z8 = {0, 0, 0, 0, 0, 0, 0, 0};
  __builtin_amdgcn_tensor_load_to_lds(g0, g1, z4, z4, z8, 0);
}
#endif

// ---------------------------------------------------------------------------
// 1) streaming f32 -> bf16 convert
// ---------------------------------------------------------------------------
__global__ void cvt_f32_bf16(const float* __restrict__ in,
                             unsigned short* __restrict__ out, size_t n) {
  size_t i = (size_t)blockIdx.x * blockDim.x + threadIdx.x;
  size_t stride = (size_t)gridDim.x * blockDim.x;
  for (; i < n; i += stride) out[i] = f2bf(in[i]);
}

// ---------------------------------------------------------------------------
// 2/4) bf16 WMMA GEMM:  C[M,N] = A[M,K] @ Bw[N,K]^T + bias[N]
//   MODE 0: store f32 row-major to outF
//   MODE 1: scatter fused-QKV columns into bf16 Q/K/V [B,H,S,hd] (S=2048,H=16)
// block = 256 threads (8 waves), tile 128x128, K step 32, TDM double-buffered
// ---------------------------------------------------------------------------
template <int MODE>
__global__ __launch_bounds__(256)
void gemm_bf16_wmma(const unsigned short* __restrict__ A,
                    const unsigned short* __restrict__ Bw,
                    const float* __restrict__ bias,
                    float* __restrict__ outF,
                    unsigned short* __restrict__ Qp,
                    unsigned short* __restrict__ Kp,
                    unsigned short* __restrict__ Vp,
                    int M, int N, int K) {
#if HAVE_TDM
  __shared__ unsigned short Asm[2][128 * 32];
  __shared__ unsigned short Bsm[2][128 * 32];
#else
  __shared__ unsigned short Asm[1][128 * 32];
  __shared__ unsigned short Bsm[1][128 * 32];
#endif

  const int tid  = threadIdx.x;
  const int lane = tid & 31;
  const int wv   = tid >> 5;       // wave 0..7
  const int half = lane >> 4;      // 0 | 1 (wave32 halves)
  const int l16  = lane & 15;
  const int bm   = blockIdx.x * 128;
  const int bn   = blockIdx.y * 128;
  const int wm   = (wv >> 1) * 32; // wave M offset within tile
  const int wn   = (wv & 1) * 64;  // wave N offset within tile

  v8f acc[2][4];
#pragma unroll
  for (int i = 0; i < 2; ++i)
#pragma unroll
    for (int j = 0; j < 4; ++j) acc[i][j] = {};

  const int nk = K >> 5;

#if HAVE_TDM
  // ---- prologue: TDM-stage tile 0 into buffer 0 (wave 0 issues) ----
  if (wv == 0) {
    tdm_load_2d_bf16((unsigned)(size_t)&Asm[0][0], A + (size_t)bm * K, 32, 128, K);
    tdm_load_2d_bf16((unsigned)(size_t)&Bsm[0][0], Bw + (size_t)bn * K, 32, 128, K);
  }
  for (int kk = 0; kk < nk; ++kk) {
    const int cur = kk & 1;
    if (wv == 0) __builtin_amdgcn_s_wait_tensorcnt(0);   // tile `cur` landed
    __syncthreads();
    if (kk + 1 < nk && wv == 0) {    // overlap next transfer with compute
      const int k1 = (kk + 1) << 5;
      tdm_load_2d_bf16((unsigned)(size_t)&Asm[1 - cur][0],
                       A + (size_t)bm * K + k1, 32, 128, K);
      tdm_load_2d_bf16((unsigned)(size_t)&Bsm[1 - cur][0],
                       Bw + (size_t)bn * K + k1, 32, 128, K);
    }
#else
  const int lr = tid >> 1;          // LDS-fill row 0..127
  const int lc = (tid & 1) * 16;    // LDS-fill col 0|16 (elements)
  for (int kk = 0; kk < nk; ++kk) {
    const int cur = 0;
    const int k0 = kk << 5;
    {
      const uint4* srcA = (const uint4*)(A + (size_t)(bm + lr) * K + k0 + lc);
      uint4* dstA = (uint4*)(&Asm[0][lr * 32 + lc]);
      dstA[0] = srcA[0]; dstA[1] = srcA[1];
      const uint4* srcB = (const uint4*)(Bw + (size_t)(bn + lr) * K + k0 + lc);
      uint4* dstB = (uint4*)(&Bsm[0][lr * 32 + lc]);
      dstB[0] = srcB[0]; dstB[1] = srcB[1];
    }
    __syncthreads();
#endif

    // ---- fragments + 8 WMMA per wave per K step ----
    FragBF af[2], bf[4];
#pragma unroll
    for (int mi = 0; mi < 2; ++mi) {
      const unsigned short* p = &Asm[cur][(wm + mi * 16 + l16) * 32 + half * 8];
      af[mi].q[0] = *(const uint4*)p;
      af[mi].q[1] = *(const uint4*)(p + 16);
    }
#pragma unroll
    for (int ni = 0; ni < 4; ++ni) {
      const unsigned short* p = &Bsm[cur][(wn + ni * 16 + l16) * 32 + half * 8];
      bf[ni].q[0] = *(const uint4*)p;
      bf[ni].q[1] = *(const uint4*)(p + 16);
    }
#pragma unroll
    for (int mi = 0; mi < 2; ++mi)
#pragma unroll
      for (int ni = 0; ni < 4; ++ni)
        acc[mi][ni] = __builtin_amdgcn_wmma_f32_16x16x32_bf16(
            false, af[mi].v, false, bf[ni].v, (short)0, acc[mi][ni],
            false, false);
    __syncthreads();   // all reads of buffer `cur` done before its reuse
  }

  // ---- epilogue ----
#pragma unroll
  for (int mi = 0; mi < 2; ++mi)
#pragma unroll
    for (int ni = 0; ni < 4; ++ni)
#pragma unroll
      for (int v = 0; v < 8; ++v) {
        const int m = bm + wm + mi * 16 + v + half * 8;   // C-layout row
        const int n = bn + wn + ni * 16 + l16;            // C-layout col
        const float val = acc[mi][ni][v] + bias[n];
        if (MODE == 0) {
          outF[(size_t)m * N + n] = val;
        } else {
          const int t = n >> 11;            // 0:q 1:k 2:v
          const int h = (n >> 7) & 15;
          const int d = n & 127;
          const int b = m >> 11;            // S = 2048
          const int s = m & 2047;
          unsigned short* p = (t == 0) ? Qp : (t == 1) ? Kp : Vp;
          p[((((size_t)b * 16 + h) << 11) + s) * 128 + d] = f2bf(val);
        }
      }
}

// ---------------------------------------------------------------------------
// 3) flash attention: one block = 128 q rows of one (b,h); 8 waves x 16 rows.
//    K tile = 64 rows (TDM-staged). O accum f32, online softmax (exp2 domain).
// ---------------------------------------------------------------------------
__global__ __launch_bounds__(256)
void attn_fwd_wmma(const unsigned short* __restrict__ Q,
                   const unsigned short* __restrict__ K,
                   const unsigned short* __restrict__ V,
                   unsigned short* __restrict__ O /* [B,S,D] bf16 */) {
  __shared__ unsigned short Ksm[64 * 128];   // [krow][hd]
  __shared__ unsigned short Vtsm[128 * 64];  // [hd][krow] (transposed)
  __shared__ unsigned short Psm[128 * 64];   // [qrow][krow]

  const int tid  = threadIdx.x;
  const int lane = tid & 31;
  const int wv   = tid >> 5;
  const int half = lane >> 4;
  const int l16  = lane & 15;
  const int qt = blockIdx.x, h = blockIdx.y, b = blockIdx.z;

  const size_t bh = (size_t)b * 16 + h;
  const unsigned short* Qb = Q + (bh << 11) * 128;
  const unsigned short* Kb = K + (bh << 11) * 128;
  const unsigned short* Vb = V + (bh << 11) * 128;

  // Q fragments for this wave's 16 rows (A layout, 4 K-steps of 32)
  const int qrow = qt * 128 + wv * 16 + l16;
  FragBF qf[4];
#pragma unroll
  for (int ks = 0; ks < 4; ++ks) {
    const unsigned short* p = Qb + (size_t)qrow * 128 + ks * 32 + half * 8;
    qf[ks].q[0] = *(const uint4*)p;
    qf[ks].q[1] = *(const uint4*)(p + 16);
  }

  v8f o[8];
#pragma unroll
  for (int j = 0; j < 8; ++j) o[j] = {};
  float mrow[8], lrow[8];
#pragma unroll
  for (int v = 0; v < 8; ++v) { mrow[v] = -1e30f; lrow[v] = 0.0f; }

  // fold 1/sqrt(128) and log2(e) into one scale; softmax runs in exp2 domain
  const float sc = 0.08838834764831845f * 1.4426950408889634f;

  for (int kt = 0; kt < 32; ++kt) {
    // ---- stage K tile: TDM (wave 0) overlapped with V-transpose (all) ----
#if HAVE_TDM
    if (wv == 0)
      tdm_load_2d_bf16((unsigned)(size_t)&Ksm[0],
                       Kb + (size_t)(kt * 64) * 128, 128, 64, 128);
#else
    {
      const int r = tid >> 2;
      const int c = (tid & 3) * 32;
      const uint4* src = (const uint4*)(Kb + (size_t)(kt * 64 + r) * 128 + c);
      uint4* dst = (uint4*)(&Ksm[r * 128 + c]);
      dst[0] = src[0]; dst[1] = src[1]; dst[2] = src[2]; dst[3] = src[3];
    }
#endif
    // ---- stage V tile transposed [hd][krow] ----
    {
      const int r = tid >> 2;
      const int c0 = (tid & 3) * 32;
#pragma unroll
      for (int cc = 0; cc < 32; cc += 8) {
        uint4 d = *(const uint4*)(Vb + (size_t)(kt * 64 + r) * 128 + c0 + cc);
        const unsigned short* e = (const unsigned short*)&d;
#pragma unroll
        for (int i = 0; i < 8; ++i) Vtsm[(c0 + cc + i) * 64 + r] = e[i];
      }
    }
    if (kt + 1 < 32) {
      __builtin_prefetch(Kb + (size_t)((kt + 1) * 64 + (tid >> 2)) * 128, 0, 1);
      __builtin_prefetch(Vb + (size_t)((kt + 1) * 64 + (tid >> 2)) * 128, 0, 1);
    }
#if HAVE_TDM
    if (wv == 0) __builtin_amdgcn_s_wait_tensorcnt(0);
#endif
    __syncthreads();

    // ---- S = Q K^T : 16 WMMA per wave ----
    v8f sacc[4];
#pragma unroll
    for (int j = 0; j < 4; ++j) sacc[j] = {};
#pragma unroll
    for (int ks = 0; ks < 4; ++ks) {
#pragma unroll
      for (int j = 0; j < 4; ++j) {
        FragBF bf;
        const unsigned short* p = &Ksm[(j * 16 + l16) * 128 + ks * 32 + half * 8];
        bf.q[0] = *(const uint4*)p;
        bf.q[1] = *(const uint4*)(p + 16);
        sacc[j] = __builtin_amdgcn_wmma_f32_16x16x32_bf16(
            false, qf[ks].v, false, bf.v, (short)0, sacc[j], false, false);
      }
    }

    // ---- online softmax (rows live in 16-lane halves; xor<=8 reductions) ----
    float corr[8];
#pragma unroll
    for (int v = 0; v < 8; ++v) {
      float mx = -1e30f;
#pragma unroll
      for (int j = 0; j < 4; ++j) mx = fmaxf(mx, sacc[j][v]);
      mx *= sc;
#pragma unroll
      for (int off = 1; off < 16; off <<= 1)
        mx = fmaxf(mx, __shfl_xor(mx, off, 32));
      const float mnew = fmaxf(mrow[v], mx);
      corr[v] = exp2f(mrow[v] - mnew);
      float rs = 0.0f;
#pragma unroll
      for (int j = 0; j < 4; ++j) {
        const float pv = exp2f(sacc[j][v] * sc - mnew);
        rs += pv;
        Psm[(wv * 16 + v + half * 8) * 64 + j * 16 + l16] = f2bf(pv);
      }
#pragma unroll
      for (int off = 1; off < 16; off <<= 1) rs += __shfl_xor(rs, off, 32);
      lrow[v] = lrow[v] * corr[v] + rs;
      mrow[v] = mnew;
    }
#pragma unroll
    for (int j = 0; j < 8; ++j)
#pragma unroll
      for (int v = 0; v < 8; ++v) o[j][v] = o[j][v] * corr[v];

    // ---- O += P V : 16 WMMA per wave (P re-read from own LDS rows) ----
#pragma unroll
    for (int ks = 0; ks < 2; ++ks) {
      FragBF pf;
      const unsigned short* pp = &Psm[(wv * 16 + l16) * 64 + ks * 32 + half * 8];
      pf.q[0] = *(const uint4*)pp;
      pf.q[1] = *(const uint4*)(pp + 16);
#pragma unroll
      for (int j = 0; j < 8; ++j) {
        FragBF vf;
        const unsigned short* vp = &Vtsm[(j * 16 + l16) * 64 + ks * 32 + half * 8];
        vf.q[0] = *(const uint4*)vp;
        vf.q[1] = *(const uint4*)(vp + 16);
        o[j] = __builtin_amdgcn_wmma_f32_16x16x32_bf16(
            false, pf.v, false, vf.v, (short)0, o[j], false, false);
      }
    }
    __syncthreads();
  }

  // ---- normalize + store bf16 [B,S,D] ----
#pragma unroll
  for (int v = 0; v < 8; ++v) {
    const float inv = 1.0f / fmaxf(lrow[v], 1e-30f);
    const int srow = qt * 128 + wv * 16 + v + half * 8;
    const size_t base = ((size_t)b * 2048 + srow) * 2048 + h * 128;
#pragma unroll
    for (int j = 0; j < 8; ++j)
      O[base + j * 16 + l16] = f2bf(o[j][v] * inv);
  }
}

// ---------------------------------------------------------------------------
// host launcher
// ---------------------------------------------------------------------------
extern "C" void kernel_launch(void* const* d_in, const int* in_sizes, int n_in,
                              void* d_out, int out_size, void* d_ws,
                              size_t ws_size, hipStream_t stream) {
  (void)in_sizes; (void)n_in; (void)out_size; (void)ws_size;

  const float* x     = (const float*)d_in[0];   // [B,S,D]
  const float* w_qkv = (const float*)d_in[1];   // [3D,D]
  const float* b_qkv = (const float*)d_in[2];   // [3D]
  const float* w_o   = (const float*)d_in[3];   // [D,D]
  const float* b_o   = (const float*)d_in[4];   // [D]
  float* out = (float*)d_out;                   // [B,S,D] f32

  const size_t XN  = (size_t)4 * 2048 * 2048;   // 16.78M
  const size_t WQN = (size_t)6144 * 2048;       // 12.58M
  const size_t WON = (size_t)2048 * 2048;       //  4.19M

  char* ws = (char*)d_ws;
  size_t off = 0;
  auto carve = [&](size_t bytes) {
    void* p = ws + off;
    off += (bytes + 255) & ~(size_t)255;
    return p;
  };
  unsigned short* xb    = (unsigned short*)carve(XN * 2);
  unsigned short* wqkvb = (unsigned short*)carve(WQN * 2);
  unsigned short* wob   = (unsigned short*)carve(WON * 2);
  unsigned short* Qb    = (unsigned short*)carve(XN * 2);
  unsigned short* Kb    = (unsigned short*)carve(XN * 2);
  unsigned short* Vb    = (unsigned short*)carve(XN * 2);
  unsigned short* attnb = (unsigned short*)carve(XN * 2);   // [B,S,D] bf16

  // 1) convert to bf16
  cvt_f32_bf16<<<2048, 256, 0, stream>>>(x, xb, XN);
  cvt_f32_bf16<<<2048, 256, 0, stream>>>(w_qkv, wqkvb, WQN);
  cvt_f32_bf16<<<1024, 256, 0, stream>>>(w_o, wob, WON);

  // 2) fused QKV projection -> Q,K,V bf16 [B,H,S,hd]
  {
    dim3 grid(8192 / 128, 6144 / 128);   // 64 x 48
    gemm_bf16_wmma<1><<<grid, 256, 0, stream>>>(
        xb, wqkvb, b_qkv, nullptr, Qb, Kb, Vb, 8192, 6144, 2048);
  }

  // 3) attention -> attn_out bf16 [B,S,D]
  {
    dim3 grid(16, 16, 4);                // qtiles x H x B
    attn_fwd_wmma<<<grid, 256, 0, stream>>>(Qb, Kb, Vb, attnb);
  }

  // 4) output projection -> f32 d_out
  {
    dim3 grid(8192 / 128, 2048 / 128);   // 64 x 16
    gemm_bf16_wmma<0><<<grid, 256, 0, stream>>>(
        attnb, wob, b_o, out, nullptr, nullptr, nullptr, 8192, 2048, 2048);
  }
}
